// Int8LlamaMLP_22565758173302
// MI455X (gfx1250) — 8-bit matrix operand order repaired and checked by reading, compiled, not run on hardware
//
#include <hip/hip_runtime.h>
#include <hip/hip_bf16.h>
#include <stdint.h>
#include <stddef.h>


typedef int v8i __attribute__((ext_vector_type(8)));

static constexpr int T_TOK = 8192;
static constexpr int HID   = 4096;
static constexpr int INTER = 11008;

static constexpr int BM = 128;
static constexpr int BN = 64;
static constexpr int BK = 64;

#define USE_ASYNC_LDS 1

__device__ __forceinline__ unsigned lds_off32(const void* p) {
  return (unsigned)(uintptr_t)p;
}

__device__ __forceinline__ void async_copy_b128(void* lds_dst, const void* gsrc) {
#if USE_ASYNC_LDS
  asm volatile("global_load_async_to_lds_b128 %0, %1, off"
               :: "v"(lds_off32(lds_dst)), "v"(gsrc)
               : "memory");
#else
  *reinterpret_cast<uint4*>(lds_dst) = *reinterpret_cast<const uint4*>(gsrc);
#endif
}

template <int N>
__device__ __forceinline__ void wait_async() {
#if USE_ASYNC_LDS
#if __has_builtin(__builtin_amdgcn_s_wait_asynccnt)
  __builtin_amdgcn_s_wait_asynccnt(N);
#else
  asm volatile("s_wait_asynccnt %0" :: "i"(N) : "memory");
#endif
#endif
}

__global__ void __launch_bounds__(256)
pack_i32_to_i8(const int* __restrict__ src, unsigned* __restrict__ dst, int n4) {
  int i = blockIdx.x * blockDim.x + threadIdx.x;
  if (i >= n4) return;
  int4 v = reinterpret_cast<const int4*>(src)[i];
  unsigned p = (unsigned)(v.x & 0xff)
             | ((unsigned)(v.y & 0xff) << 8)
             | ((unsigned)(v.z & 0xff) << 16)
             | ((unsigned)(v.w & 0xff) << 24);
  dst[i] = p;
}

__global__ void __launch_bounds__(256)
gateup_kernel(const signed char* __restrict__ X,
              const signed char* __restrict__ GW,
              const signed char* __restrict__ UW,
              const float* __restrict__ gate_a_p,
              const float* __restrict__ gate_b,
              const float* __restrict__ up_a_p,
              const float* __restrict__ up_b,
              signed char* __restrict__ Q) {
  constexpr int STAGE_DW = (BM * BK + 2 * BN * BK) / 4;
  __shared__ unsigned smem[2 * STAGE_DW];

  const int t    = threadIdx.x;
  const int lane = t & 31;
  const int wave = t >> 5;
  const int wm   = wave >> 1;
  const int wn   = wave & 1;
  const int hi   = lane >> 4;
  const int l15  = lane & 15;

  const int m0 = blockIdx.y * BM;
  const int n0 = blockIdx.x * BN;

  const int srow = t >> 2;
  const int scol = (t & 3) * 16;

  v8i accg[2][2] = {};
  v8i accu[2][2] = {};

  auto issue_stage = [&](int kk, int s) {
    const int k0 = kk * BK;
    unsigned* base = &smem[s * STAGE_DW];
    async_copy_b128(&base[(srow)      * 16 + (scol >> 2)],
                    X + (size_t)(m0 + srow)      * HID + k0 + scol);
    async_copy_b128(&base[(srow + 64) * 16 + (scol >> 2)],
                    X + (size_t)(m0 + srow + 64) * HID + k0 + scol);
    async_copy_b128(&base[2048 + srow * 16 + (scol >> 2)],
                    GW + (size_t)(n0 + srow) * HID + k0 + scol);
    async_copy_b128(&base[3072 + srow * 16 + (scol >> 2)],
                    UW + (size_t)(n0 + srow) * HID + k0 + scol);
  };

  issue_stage(0, 0);

  constexpr int KSTEPS = HID / BK;
  for (int kk = 0; kk < KSTEPS; ++kk) {
    if (kk + 1 < KSTEPS) {
      issue_stage(kk + 1, (kk + 1) & 1);
      wait_async<4>();
    } else {
      wait_async<0>();
    }
    __syncthreads();

    const unsigned* base = &smem[(kk & 1) * STAGE_DW];

    v8i af[2], bg[2], bu[2];
#pragma unroll
    for (int i = 0; i < 2; ++i) {
      const int row = wm * 32 + i * 16 + l15;
#pragma unroll
      for (int v = 0; v < 8; ++v)
        af[i][v] = (int)base[row * 16 + ((v >> 2) << 3) + (hi << 2) + (v & 3)];
    }
#pragma unroll
    for (int j = 0; j < 2; ++j) {
      const int col = wn * 32 + j * 16 + l15;
#pragma unroll
      for (int v = 0; v < 8; ++v) {
        const int o = col * 16 + ((v >> 2) << 3) + (hi << 2) + (v & 3);
        bg[j][v] = (int)base[2048 + o];
        bu[j][v] = (int)base[3072 + o];
      }
    }

#pragma unroll
    for (int i = 0; i < 2; ++i)
#pragma unroll
      for (int j = 0; j < 2; ++j) {
        accg[i][j] = __builtin_amdgcn_wmma_i32_16x16x64_iu8(
            true, af[i], true, bg[j], accg[i][j], false, false);
        accu[i][j] = __builtin_amdgcn_wmma_i32_16x16x64_iu8(
            true, af[i], true, bu[j], accu[i][j], false, false);
      }
    __syncthreads();
  }

  const float ga = *gate_a_p;
  const float ua = *up_a_p;
#pragma unroll
  for (int i = 0; i < 2; ++i) {
#pragma unroll
    for (int j = 0; j < 2; ++j) {
      const int n  = n0 + wn * 32 + j * 16 + l15;
      const int mb = m0 + wm * 32 + i * 16 + hi * 8;
      const float gb = gate_b[n];
      const float ub = up_b[n];
#pragma unroll
      for (int r = 0; r < 8; ++r) {
        float g = (float)accg[i][j][r] * ga + gb;
        float s = g / (1.0f + __expf(-g));
        float u = (float)accu[i][j][r] * ua + ub;
        float x = s * u;
        float q = fminf(fmaxf(rintf(x), -128.0f), 127.0f);
        Q[(size_t)(mb + r) * INTER + n] = (signed char)(int)q;
      }
    }
  }
}

__global__ void __launch_bounds__(256)
down_kernel(const signed char* __restrict__ Q,
            const signed char* __restrict__ DW,
            const float* __restrict__ down_a_p,
            const float* __restrict__ down_b,
            float* __restrict__ OUT) {
  constexpr int STAGE_DW = (BM * BK + BN * BK) / 4;
  __shared__ unsigned smem[2 * STAGE_DW];

  const int t    = threadIdx.x;
  const int lane = t & 31;
  const int wave = t >> 5;
  const int wm   = wave >> 1;
  const int wn   = wave & 1;
  const int hi   = lane >> 4;
  const int l15  = lane & 15;

  const int m0 = blockIdx.y * BM;
  const int n0 = blockIdx.x * BN;

  const int srow = t >> 2;
  const int scol = (t & 3) * 16;

  v8i acc[2][2] = {};

  auto issue_stage = [&](int kk, int s) {
    const int k0 = kk * BK;
    unsigned* base = &smem[s * STAGE_DW];
    async_copy_b128(&base[(srow)      * 16 + (scol >> 2)],
                    Q + (size_t)(m0 + srow)      * INTER + k0 + scol);
    async_copy_b128(&base[(srow + 64) * 16 + (scol >> 2)],
                    Q + (size_t)(m0 + srow + 64) * INTER + k0 + scol);
    async_copy_b128(&base[2048 + srow * 16 + (scol >> 2)],
                    DW + (size_t)(n0 + srow) * INTER + k0 + scol);
  };

  issue_stage(0, 0);

  constexpr int KSTEPS = INTER / BK;
  for (int kk = 0; kk < KSTEPS; ++kk) {
    if (kk + 1 < KSTEPS) {
      issue_stage(kk + 1, (kk + 1) & 1);
      wait_async<3>();
    } else {
      wait_async<0>();
    }
    __syncthreads();

    const unsigned* base = &smem[(kk & 1) * STAGE_DW];

    v8i af[2], bf[2];
#pragma unroll
    for (int i = 0; i < 2; ++i) {
      const int row = wm * 32 + i * 16 + l15;
#pragma unroll
      for (int v = 0; v < 8; ++v)
        af[i][v] = (int)base[row * 16 + ((v >> 2) << 3) + (hi << 2) + (v & 3)];
    }
#pragma unroll
    for (int j = 0; j < 2; ++j) {
      const int col = wn * 32 + j * 16 + l15;
#pragma unroll
      for (int v = 0; v < 8; ++v)
        bf[j][v] = (int)base[2048 + col * 16 + ((v >> 2) << 3) + (hi << 2) + (v & 3)];
    }

#pragma unroll
    for (int i = 0; i < 2; ++i)
#pragma unroll
      for (int j = 0; j < 2; ++j)
        acc[i][j] = __builtin_amdgcn_wmma_i32_16x16x64_iu8(
            true, af[i], true, bf[j], acc[i][j], false, false);
    __syncthreads();
  }

  const float da = *down_a_p;
#pragma unroll
  for (int i = 0; i < 2; ++i) {
#pragma unroll
    for (int j = 0; j < 2; ++j) {
      const int n  = n0 + wn * 32 + j * 16 + l15;
      const int mb = m0 + wm * 32 + i * 16 + hi * 8;
      const float db = down_b[n];
#pragma unroll
      for (int r = 0; r < 8; ++r)
        OUT[(size_t)(mb + r) * HID + n] = (float)acc[i][j][r] * da + db;
    }
  }
}

extern "C" void kernel_launch(void* const* d_in, const int* in_sizes, int n_in,
                              void* d_out, int out_size, void* d_ws, size_t ws_size,
                              hipStream_t stream) {
  (void)in_sizes; (void)n_in; (void)out_size; (void)ws_size;

  const int*   hs = (const int*)  d_in[0];
  const int*   gw = (const int*)  d_in[1];
  const float* ga = (const float*)d_in[2];
  const float* gb = (const float*)d_in[3];
  const int*   uw = (const int*)  d_in[4];
  const float* ua = (const float*)d_in[5];
  const float* ub = (const float*)d_in[6];
  const int*   dw = (const int*)  d_in[7];
  const float* da = (const float*)d_in[8];
  const float* db = (const float*)d_in[9];
  float* out = (float*)d_out;

  signed char* X8 = (signed char*)d_ws;
  signed char* G8 = X8 + (size_t)T_TOK * HID;
  signed char* U8 = G8 + (size_t)INTER * HID;
  signed char* D8 = U8 + (size_t)INTER * HID;
  signed char* Q8 = D8 + (size_t)HID * INTER;

  const int n4x = T_TOK * HID / 4;
  const int n4w = INTER * HID / 4;
  pack_i32_to_i8<<<(n4x + 255) / 256, 256, 0, stream>>>(hs, (unsigned*)X8, n4x);
  pack_i32_to_i8<<<(n4w + 255) / 256, 256, 0, stream>>>(gw, (unsigned*)G8, n4w);
  pack_i32_to_i8<<<(n4w + 255) / 256, 256, 0, stream>>>(uw, (unsigned*)U8, n4w);
  pack_i32_to_i8<<<(n4w + 255) / 256, 256, 0, stream>>>(dw, (unsigned*)D8, n4w);

  dim3 g1(INTER / BN, T_TOK / BM);
  gateup_kernel<<<g1, 256, 0, stream>>>(X8, G8, U8, ga, gb, ua, ub, Q8);

  dim3 g2(HID / BN, T_TOK / BM);
  down_kernel<<<g2, 256, 0, stream>>>(Q8, D8, da, db, out);
}
